// SingleHeadAttention_54176717472071
// MI455X (gfx1250) — matrix operand order repaired and checked by reading, not compiled, not run on hardware
//
#include <hip/hip_runtime.h>

// ---------------------------------------------------------------------------
// SingleHeadAttention (causal), B=8, S=2048, D=DK=DV=1024, fp32 in/out.
// bf16 WMMA (f32 accumulate) for all matmuls, fp32 softmax.
// Wave tile 32x64 (8 wmma / 12 b128-loads per K-step), block tile 128x128.
// Causal-exact work skipping in both attention GEMMs.
// ---------------------------------------------------------------------------

typedef __bf16 bf16x16 __attribute__((ext_vector_type(16)));
typedef float f32x8 __attribute__((ext_vector_type(8)));

union Frag {
    bf16x16 v;
    uint4 u[2];
};

__device__ __forceinline__ unsigned short f2bf(float f) {
    union { float f; unsigned u; } cv;
    cv.f = f;
    unsigned u = cv.u;
    unsigned r = u + 0x7FFFu + ((u >> 16) & 1u);   // round-to-nearest-even
    return (unsigned short)(r >> 16);
}

// ------------------------- conversion kernels ------------------------------

__global__ __launch_bounds__(256) void sha_cvt_f32_bf16(
    const float* __restrict__ src, unsigned short* __restrict__ dst, long long n) {
    long long i = (long long)blockIdx.x * 256 + threadIdx.x;
    if (i < n) dst[i] = f2bf(src[i]);
}

// W [1024 x 1024] row-major -> WT [1024 x 1024] row-major (transposed), bf16
__global__ __launch_bounds__(256) void sha_cvt_wT(
    const float* __restrict__ W, unsigned short* __restrict__ WT) {
    int i = blockIdx.x * 256 + threadIdx.x;     // 0 .. 1M-1
    int n = i >> 10;
    int k = i & 1023;
    WT[(long long)n * 1024 + k] = f2bf(W[(long long)k * 1024 + n]);
}

// ------------------------- WMMA GEMM kernel --------------------------------
// C[M,N] = A[M,K] * Bt[N,K]^T  (Bt row n is column n of the math B matrix).
// Block: 256 threads = 8 waves, waves tiled 4(M) x 2(N), wave tile 32x64,
// block tile 128x128, K step 32 (v_wmma_f32_16x16x32_bf16).
//
// MODE 0: store bf16 row-major [m*ldo + n]          (Q, K projections)
// MODE 1: store bf16 transposed into Vt[B, DV, S]   (V projection)
// MODE 2: store f32 scaled, causal block-skip       (scores, per-batch z)
// MODE 3: store f32 row-major, causal K-trim        (attention output)

template <int MODE>
__global__ __launch_bounds__(256) void sha_gemm_bf16_wmma(
    const unsigned short* __restrict__ A,
    const unsigned short* __restrict__ Bt,
    void* __restrict__ OutRaw,
    int Kdim, int lda, int ldb, int ldo,
    long long sA, long long sB, long long sO, float scale) {
    const int b = blockIdx.z;
    A  += (long long)b * sA;
    Bt += (long long)b * sB;

    const int m_blk = blockIdx.y * 128;
    const int n_blk = blockIdx.x * 128;
    if (MODE == 2 && n_blk > m_blk + 127) return;  // fully causally-masked block

    // P[m,k]==0 for k>m, so the P*V GEMM needs only k < m_blk+128 (exact).
    int Klim = Kdim;
    if (MODE == 3) Klim = (Kdim < m_blk + 128) ? Kdim : (m_blk + 128);

    const int tid  = threadIdx.x;
    const int lane = tid & 31;
    const int wave = tid >> 5;
    const int m0 = m_blk + (wave >> 1) * 32;
    const int n0 = n_blk + (wave & 1) * 64;
    const int lg = lane >> 4;     // lane half-group (0: lanes 0-15, 1: 16-31)
    const int lm = lane & 15;

    // A fragment (16x32 bf16): lane = row, lg selects K 0-7/16-23 vs 8-15/24-31
    const unsigned short* arow0 = A + (long long)(m0 + lm) * lda;
    const unsigned short* arow1 = A + (long long)(m0 + 16 + lm) * lda;
    // B fragment (32x16 bf16): lane = col, lg selects K 0-7/16-23 vs 8-15/24-31
    const unsigned short* brow[4];
#pragma unroll
    for (int j = 0; j < 4; ++j)
        brow[j] = Bt + (long long)(n0 + 16 * j + lm) * ldb;
    const int aoff = lg * 8;
    const int boff = lg * 8;

    f32x8 c[2][4];
#pragma unroll
    for (int i = 0; i < 2; ++i)
#pragma unroll
        for (int j = 0; j < 4; ++j) c[i][j] = f32x8{};

    for (int k0 = 0; k0 < Klim; k0 += 32) {
        Frag a0, a1, bf[4];
        a0.u[0] = *(const uint4*)(arow0 + k0 + aoff);
        a0.u[1] = *(const uint4*)(arow0 + k0 + 16 + aoff);
        a1.u[0] = *(const uint4*)(arow1 + k0 + aoff);
        a1.u[1] = *(const uint4*)(arow1 + k0 + 16 + aoff);
#pragma unroll
        for (int j = 0; j < 4; ++j) {
            bf[j].u[0] = *(const uint4*)(brow[j] + k0 + boff);
            bf[j].u[1] = *(const uint4*)(brow[j] + k0 + 16 + boff);
        }
#pragma unroll
        for (int j = 0; j < 4; ++j) {
            c[0][j] = __builtin_amdgcn_wmma_f32_16x16x32_bf16(
                false, a0.v, false, bf[j].v, (short)0, c[0][j], false, false);
            c[1][j] = __builtin_amdgcn_wmma_f32_16x16x32_bf16(
                false, a1.v, false, bf[j].v, (short)0, c[1][j], false, false);
        }
    }

    // C/D layout: VGPR r -> M = r + 8*lg, N = lane%16
    auto store_tile = [&](const f32x8& ct, int mb, int nb) {
        const int n = nb + lm;
#pragma unroll
        for (int r = 0; r < 8; ++r) {
            const int m = mb + r + 8 * lg;
            const float val = ct[r];
            if (MODE == 0) {
                unsigned short* out = (unsigned short*)OutRaw;
                out[(long long)m * ldo + n] = f2bf(val);
            } else if (MODE == 1) {
                // Vt[B, DV=1024, S=2048]; global row m = b*2048 + s
                unsigned short* out = (unsigned short*)OutRaw;
                const long long bb = m >> 11;
                const long long s  = m & 2047;
                out[bb * (1024LL * 2048LL) + (long long)n * 2048LL + s] = f2bf(val);
            } else if (MODE == 2) {
                float* out = (float*)OutRaw + (long long)b * sO;
                out[(long long)m * ldo + n] = val * scale;
            } else {
                float* out = (float*)OutRaw + (long long)b * sO;
                out[(long long)m * ldo + n] = val;
            }
        }
    };
#pragma unroll
    for (int i = 0; i < 2; ++i)
#pragma unroll
        for (int j = 0; j < 4; ++j)
            store_tile(c[i][j], m0 + 16 * i, n0 + 16 * j);
}

// ------------------------- causal softmax ----------------------------------
// One 256-thread block per (b, q) row. Reads fp32 scores (valid cols 0..m),
// writes bf16 probabilities with zeros in the masked tail.

__global__ __launch_bounds__(256) void sha_softmax_causal(
    const float* __restrict__ scores, unsigned short* __restrict__ P) {
    const int row = blockIdx.x;                  // 0 .. 16383  (= b*2048 + m)
    const int m = row & 2047;
    const int len = m + 1;
    const float* srow = scores + (long long)row * 2048;
    unsigned short* prow = P + (long long)row * 2048;
    const int tid = threadIdx.x;

    __shared__ float redmax[8];
    __shared__ float redsum[8];

    float mx = -3.4028235e38f;
    for (int i = tid; i < len; i += 256) mx = fmaxf(mx, srow[i]);
#pragma unroll
    for (int off = 16; off > 0; off >>= 1) mx = fmaxf(mx, __shfl_xor(mx, off, 32));
    if ((tid & 31) == 0) redmax[tid >> 5] = mx;
    __syncthreads();
    mx = redmax[0];
#pragma unroll
    for (int w = 1; w < 8; ++w) mx = fmaxf(mx, redmax[w]);

    float sum = 0.0f;
    for (int i = tid; i < len; i += 256) sum += __expf(srow[i] - mx);
#pragma unroll
    for (int off = 16; off > 0; off >>= 1) sum += __shfl_xor(sum, off, 32);
    if ((tid & 31) == 0) redsum[tid >> 5] = sum;
    __syncthreads();
    sum = 0.0f;
#pragma unroll
    for (int w = 0; w < 8; ++w) sum += redsum[w];
    const float inv = 1.0f / sum;

    for (int i = tid; i < 2048; i += 256) {
        const float p = (i < len) ? __expf(srow[i] - mx) * inv : 0.0f;
        prow[i] = f2bf(p);
    }
}

// ------------------------- launch ------------------------------------------

extern "C" void kernel_launch(void* const* d_in, const int* in_sizes, int n_in,
                              void* d_out, int out_size, void* d_ws, size_t ws_size,
                              hipStream_t stream) {
    (void)in_sizes; (void)n_in; (void)out_size; (void)ws_size;

    const float* x  = (const float*)d_in[0];   // [8, 2048, 1024]
    const float* Wq = (const float*)d_in[1];   // [1024, 1024]
    const float* Wk = (const float*)d_in[2];
    const float* Wv = (const float*)d_in[3];
    float* out = (float*)d_out;                // [8, 2048, 1024]

    // Workspace layout
    char* ws = (char*)d_ws;
    unsigned short* xb  = (unsigned short*)ws;  ws += 16384LL * 1024 * 2;   // x bf16
    unsigned short* WqT = (unsigned short*)ws;  ws += 1024LL * 1024 * 2;    // Wq^T bf16
    unsigned short* WkT = (unsigned short*)ws;  ws += 1024LL * 1024 * 2;
    unsigned short* WvT = (unsigned short*)ws;  ws += 1024LL * 1024 * 2;
    unsigned short* Qb  = (unsigned short*)ws;  ws += 16384LL * 1024 * 2;   // Q bf16 [B*S, DK]
    unsigned short* Kb  = (unsigned short*)ws;  ws += 16384LL * 1024 * 2;   // K bf16 [B*S, DK]
    unsigned short* Vt  = (unsigned short*)ws;  ws += 8LL * 1024 * 2048 * 2;// V^T bf16 [B, DV, S]
    float*          Sc  = (float*)ws;           ws += 8LL * 2048 * 2048 * 4;// scores f32 [B, S, S]
    unsigned short* P   = (unsigned short*)ws;                              // probs bf16 [B, S, S]

    // 1) convert inputs to bf16 (weights transposed)
    sha_cvt_f32_bf16<<<65536, 256, 0, stream>>>(x, xb, 16384LL * 1024);
    sha_cvt_wT<<<4096, 256, 0, stream>>>(Wq, WqT);
    sha_cvt_wT<<<4096, 256, 0, stream>>>(Wk, WkT);
    sha_cvt_wT<<<4096, 256, 0, stream>>>(Wv, WvT);

    // 2) projections: M=16384, N=1024, K=1024   (block tile 128x128)
    dim3 gProj(1024 / 128, 16384 / 128, 1);
    sha_gemm_bf16_wmma<0><<<gProj, 256, 0, stream>>>(xb, WqT, Qb, 1024, 1024, 1024, 1024,
                                                     0, 0, 0, 1.0f);
    sha_gemm_bf16_wmma<0><<<gProj, 256, 0, stream>>>(xb, WkT, Kb, 1024, 1024, 1024, 1024,
                                                     0, 0, 0, 1.0f);
    sha_gemm_bf16_wmma<1><<<gProj, 256, 0, stream>>>(xb, WvT, Vt, 1024, 1024, 1024, 0,
                                                     0, 0, 0, 1.0f);

    // 3) scores = (Q K^T) / sqrt(1024), per batch; above-diagonal blocks skipped
    dim3 gScore(2048 / 128, 2048 / 128, 8);
    sha_gemm_bf16_wmma<2><<<gScore, 256, 0, stream>>>(Qb, Kb, Sc, 1024, 1024, 1024, 2048,
                                                      2048LL * 1024, 2048LL * 1024,
                                                      2048LL * 2048, 0.03125f);

    // 4) causal softmax (fp32), writes bf16 P with zeroed masked tail
    sha_softmax_causal<<<16384, 256, 0, stream>>>(Sc, P);

    // 5) out = P V, per batch: M=2048, N=1024, K=2048 (K trimmed to m_blk+128)
    dim3 gOut(1024 / 128, 2048 / 128, 8);
    sha_gemm_bf16_wmma<3><<<gOut, 256, 0, stream>>>(P, Vt, out, 2048, 2048, 2048, 1024,
                                                    2048LL * 2048, 1024LL * 2048,
                                                    2048LL * 1024, 1.0f);
}
